// AttentionLayer_6279242187050
// MI455X (gfx1250) — matrix operand order repaired and checked by reading, not compiled, not run on hardware
//
#include <hip/hip_runtime.h>
#include <hip/hip_bf16.h>

typedef __attribute__((ext_vector_type(16))) __bf16 v16bf;
typedef __attribute__((ext_vector_type(8)))  __bf16 bf16x8;
typedef __attribute__((ext_vector_type(8)))  float  v8f;
typedef __attribute__((ext_vector_type(4)))  float  f32x4;

#define D_MODEL 1024
#define NHEAD   16
#define HDIM    64
#define SEQ     2048
#define BATCH   4
#define M_TOTAL (BATCH * SEQ)   // 8192 token rows

#define WMMA_BF16(A, B, C) \
  __builtin_amdgcn_wmma_f32_16x16x32_bf16(false, (A), false, (B), (short)0, (C), false, false)

static __device__ __forceinline__ v8f v8f_zero() {
  v8f z;
#pragma unroll
  for (int i = 0; i < 8; ++i) z[i] = 0.0f;
  return z;
}

// A fragment (16x32 bf16): lane L -> row L&15; element e -> k=k0+16*(L>>4)+e.
// Two contiguous 16B chunks.
static __device__ __forceinline__ v16bf a_frag_bf16(const __bf16* __restrict__ row,
                                                    int half, int k0) {
  bf16x8 c0 = *(const bf16x8*)(row + k0 + 16 * half);
  bf16x8 c1 = *(const bf16x8*)(row + k0 + 16 * half + 8);
  v16bf r;
#pragma unroll
  for (int i = 0; i < 8; ++i) { r[i] = c0[i]; r[8 + i] = c1[i]; }
  return r;
}

// ---------------------------------------------------------------------------
// Stage 0: one-time f32 -> bf16 conversion (X and weights), so GEMM inner
// loops are pure bf16 loads + WMMA (no per-tile redundant cvt work).
// ---------------------------------------------------------------------------
__global__ __launch_bounds__(256) void cvt_bf16_kernel(const float* __restrict__ src,
                                                       __bf16* __restrict__ dst,
                                                       int n) {
  int i = (blockIdx.x * 256 + threadIdx.x) * 8;
  if (i >= n) return;
  f32x4 a = *(const f32x4*)(src + i);
  f32x4 b = *(const f32x4*)(src + i + 4);
  bf16x8 o;
#pragma unroll
  for (int j = 0; j < 4; ++j) { o[j] = (__bf16)a[j]; o[4 + j] = (__bf16)b[j]; }
  *(bf16x8*)(dst + i) = o;
}

// ---------------------------------------------------------------------------
// GEMM helpers: load the 2 A-frags + 4 B-frags for one 32-deep K chunk,
// and issue the 8 WMMAs for one chunk.
// ---------------------------------------------------------------------------
static __device__ __forceinline__ void gemm_load_chunk(
    const __bf16* __restrict__ arow0, const __bf16* __restrict__ arow1,
    const __bf16* __restrict__ wrow, int half, int k0,
    v16bf& a0, v16bf& a1, v16bf b[4]) {
  a0 = a_frag_bf16(arow0, half, k0);
  a1 = a_frag_bf16(arow1, half, k0);
#pragma unroll
  for (int t = 0; t < 4; ++t)
    b[t] = *(const v16bf*)(wrow + (size_t)(16 * t) * D_MODEL + k0);
}

static __device__ __forceinline__ void gemm_compute_chunk(
    const v16bf& a0, const v16bf& a1, const v16bf b[4], v8f acc[2][4]) {
#pragma unroll
  for (int t = 0; t < 4; ++t) {
    acc[0][t] = WMMA_BF16(a0, b[t], acc[0][t]);
    acc[1][t] = WMMA_BF16(a1, b[t], acc[1][t]);
  }
}

// ---------------------------------------------------------------------------
// Stage 1: projection GEMM (all-bf16, f32 accum), 32x64 tile per wave
// (2 M-subtiles x 4 N-tiles = 8 accumulators). Ping-pong register buffers
// (unroll-by-2): no rotation copies, loads never overwrite in-flight WMMA
// operands (avoids WMMA->VALU WAR nops), next chunk always in flight.
// transpose_out==0: ws[b,h,s,hd] (Q,K); ==1: ws[b,h,hd,s] (V transposed).
// ---------------------------------------------------------------------------
__global__ __launch_bounds__(128) void proj_gemm_kernel(
    const __bf16* __restrict__ X, const __bf16* __restrict__ W,
    const float* __restrict__ bias, __bf16* __restrict__ out,
    int transpose_out) {
  const int lane = threadIdx.x & 31;
  const int wave = threadIdx.x >> 5;
  const int gw   = blockIdx.x * 4 + wave;       // 4096 waves total
  const int m0   = (gw >> 4) * 32;              // 256 row tiles of 32
  const int n0   = (gw & 15) * 64;              // 16 col groups of 64
  const int half = lane >> 4;
  const int l15  = lane & 15;

  const __bf16* arow0 = X + (size_t)(m0 + l15) * D_MODEL;
  const __bf16* arow1 = arow0 + 16 * D_MODEL;
  const __bf16* wrow  = W + (size_t)(n0 + l15) * D_MODEL + 16 * half;

  v8f acc[2][4];
#pragma unroll
  for (int mi = 0; mi < 2; ++mi)
#pragma unroll
    for (int t = 0; t < 4; ++t) acc[mi][t] = v8f_zero();

  v16bf a0e, a1e, be[4];                        // even-chunk buffers
  v16bf a0o, a1o, bo[4];                        // odd-chunk buffers
  gemm_load_chunk(arow0, arow1, wrow, half, 0, a0e, a1e, be);

  for (int k0 = 0; k0 < D_MODEL; k0 += 64) {
    __builtin_prefetch(arow0 + k0 + 64, 0, 3);
    __builtin_prefetch(arow1 + k0 + 64, 0, 3);
    gemm_load_chunk(arow0, arow1, wrow, half, k0 + 32, a0o, a1o, bo);
    gemm_compute_chunk(a0e, a1e, be, acc);
    if (k0 + 64 < D_MODEL)
      gemm_load_chunk(arow0, arow1, wrow, half, k0 + 64, a0e, a1e, be);
    gemm_compute_chunk(a0o, a1o, bo, acc);
  }

  const int b = m0 / SEQ;                        // 32 | SEQ, same batch
  const int h = n0 / HDIM;                       // n0 is 64-aligned
#pragma unroll
  for (int mi = 0; mi < 2; ++mi) {
    const int s_base = (m0 % SEQ) + 16 * mi + 8 * half;
#pragma unroll
    for (int t = 0; t < 4; ++t) {
      const int hd = 16 * t + l15;
      const float bv = bias[n0 + hd];
      if (!transpose_out) {
        __bf16* op = out + (size_t)(b * NHEAD + h) * SEQ * HDIM;
#pragma unroll
        for (int r = 0; r < 8; ++r)
          op[(size_t)(s_base + r) * HDIM + hd] = (__bf16)(acc[mi][t][r] + bv);
      } else {
        bf16x8 pk;
#pragma unroll
        for (int r = 0; r < 8; ++r) pk[r] = (__bf16)(acc[mi][t][r] + bv);
        __bf16* op = out + ((size_t)(b * NHEAD + h) * HDIM + hd) * SEQ + s_base;
        *(bf16x8*)op = pk;                       // contiguous 16B store
      }
    }
  }
}

// ---------------------------------------------------------------------------
// Stage 2: flash attention. One wave owns 16 query rows of one (b,h),
// streams K/V in 32-token chunks with online softmax, f32 accumulators.
// Ping-pong K/V buffers: chunk i+1 global loads overlap chunk i's
// softmax VALU + WMMA work.
// ---------------------------------------------------------------------------
static __device__ __forceinline__ void load_k_frags(const __bf16* __restrict__ kbase,
                                                    int t0, int l15, int half,
                                                    v16bf kb[4]) {
  const __bf16* kr0 = kbase + (size_t)(t0 + l15) * HDIM + 16 * half;
  const __bf16* kr1 = kr0 + 16 * HDIM;
  kb[0] = *(const v16bf*)(kr0);
  kb[1] = *(const v16bf*)(kr0 + 32);
  kb[2] = *(const v16bf*)(kr1);
  kb[3] = *(const v16bf*)(kr1 + 32);
}

static __device__ __forceinline__ void load_v_frags(const __bf16* __restrict__ vbase,
                                                    int t0, int l15, int half,
                                                    v16bf vb[4]) {
#pragma unroll
  for (int t = 0; t < 4; ++t)
    vb[t] = *(const v16bf*)(vbase + (size_t)(16 * t + l15) * SEQ + t0 + 16 * half);
}

// One 32-token chunk: S = Q·K^T, LDS transpose, online softmax, O += P·V.
static __device__ __forceinline__ void attn_chunk(
    float* __restrict__ sS, const v16bf& qa0, const v16bf& qa1,
    const v16bf kb[4], const v16bf vb[4], v8f o[4],
    float& m_run, float& l_run, int l15, int half) {
  const float scale = 0.125f;                    // 1/sqrt(64)

  v8f s0 = v8f_zero(), s1 = v8f_zero();
  s0 = WMMA_BF16(qa0, kb[0], s0);
  s0 = WMMA_BF16(qa1, kb[1], s0);
  s1 = WMMA_BF16(qa0, kb[2], s1);
  s1 = WMMA_BF16(qa1, kb[3], s1);

  // D-layout -> LDS (row-major 16x32), scaled.
#pragma unroll
  for (int r = 0; r < 8; ++r) {
    sS[(r + 8 * half) * 32 + l15]      = s0[r] * scale;
    sS[(r + 8 * half) * 32 + 16 + l15] = s1[r] * scale;
  }
  // (DS ops are in-order within a wave; per-wave LDS region, no barrier)

  // Read back in A-operand order: lane row = l15, k chunks.
  float sv[16];
  {
    f32x4 r0 = *(const f32x4*)(sS + l15 * 32 + 16 * half);
    f32x4 r1 = *(const f32x4*)(sS + l15 * 32 + 16 * half + 4);
    f32x4 r2 = *(const f32x4*)(sS + l15 * 32 + 16 * half + 8);
    f32x4 r3 = *(const f32x4*)(sS + l15 * 32 + 16 * half + 12);
#pragma unroll
    for (int i = 0; i < 4; ++i) {
      sv[i] = r0[i]; sv[4 + i] = r1[i]; sv[8 + i] = r2[i]; sv[12 + i] = r3[i];
    }
  }

  // Online softmax: lanes (L, L+16) jointly own row l15.
  float mx = sv[0];
#pragma unroll
  for (int i = 1; i < 16; ++i) mx = fmaxf(mx, sv[i]);
  mx = fmaxf(mx, __shfl_xor(mx, 16, 32));
  const float m_new = fmaxf(m_run, mx);
  const float alpha = __expf(m_run - m_new);

  float psum = 0.0f;
  v16bf pa;
#pragma unroll
  for (int i = 0; i < 16; ++i) {
    float e = __expf(sv[i] - m_new);
    psum += e;
    pa[i] = (__bf16)e;
  }
  psum += __shfl_xor(psum, 16, 32);
  l_run = l_run * alpha + psum;
  m_run = m_new;

  // Rescale O (per-row alpha broadcast via lane shuffle).
#pragma unroll
  for (int r = 0; r < 8; ++r) {
    float ar = __shfl(alpha, r + 8 * half, 32);
#pragma unroll
    for (int t = 0; t < 4; ++t) o[t][r] *= ar;
  }

  // O += P · V (V^T layout makes the B-frag a contiguous load).
#pragma unroll
  for (int t = 0; t < 4; ++t) o[t] = WMMA_BF16(pa, vb[t], o[t]);
}

__global__ __launch_bounds__(128) void attn_kernel(
    const __bf16* __restrict__ Q,   // [B*H, SEQ, HDIM]
    const __bf16* __restrict__ K,   // [B*H, SEQ, HDIM]
    const __bf16* __restrict__ Vt,  // [B*H, HDIM, SEQ]
    __bf16* __restrict__ O) {       // [B, SEQ, D_MODEL] head-merged
  __shared__ __attribute__((aligned(16))) float lds[4 * 16 * 32];
  const int lane = threadIdx.x & 31;
  const int wave = threadIdx.x >> 5;
  float* sS = lds + wave * 512;                  // per-wave 16x32 f32 tile
  const int gw = blockIdx.x * 4 + wave;          // 8192 waves
  const int bh = gw >> 7;                        // 64 (b,h) pairs
  const int q0 = (gw & 127) * 16;                // 128 q tiles per head
  const int half = lane >> 4;
  const int l15  = lane & 15;

  const __bf16* kbase = K + (size_t)bh * SEQ * HDIM;
  const __bf16* vbase = Vt + (size_t)bh * HDIM * SEQ;

  // Q fragments are loop-invariant: preload both K-depth halves.
  const __bf16* qrow = Q + ((size_t)bh * SEQ + q0 + l15) * HDIM;
  v16bf qa0 = a_frag_bf16(qrow, half, 0);
  v16bf qa1 = a_frag_bf16(qrow, half, 32);

  v8f o[4];
#pragma unroll
  for (int t = 0; t < 4; ++t) o[t] = v8f_zero();
  float m_run = -3.0e38f;
  float l_run = 0.0f;

  // Ping-pong K/V fragment buffers (no rotation copies).
  v16bf kbe[4], vbe[4], kbo[4], vbo[4];
  load_k_frags(kbase, 0, l15, half, kbe);
  load_v_frags(vbase, 0, l15, half, vbe);

  for (int t0 = 0; t0 < SEQ; t0 += 64) {
    load_k_frags(kbase, t0 + 32, l15, half, kbo);
    load_v_frags(vbase, t0 + 32, l15, half, vbo);
    attn_chunk(sS, qa0, qa1, kbe, vbe, o, m_run, l_run, l15, half);
    if (t0 + 64 < SEQ) {
      load_k_frags(kbase, t0 + 64, l15, half, kbe);
      load_v_frags(vbase, t0 + 64, l15, half, vbe);
    }
    attn_chunk(sS, qa0, qa1, kbo, vbo, o, m_run, l_run, l15, half);
  }

  // Finalize: divide by row sums, store head-merged bf16 [b, s, D].
  const float linv = 1.0f / l_run;
  const int b = bh / NHEAD, h = bh % NHEAD;
  __bf16* obase = O + (size_t)b * SEQ * D_MODEL + (size_t)h * HDIM;
#pragma unroll
  for (int r = 0; r < 8; ++r) {
    float li = __shfl(linv, r + 8 * half, 32);
    const size_t srow = (size_t)(q0 + r + 8 * half);
#pragma unroll
    for (int t = 0; t < 4; ++t)
      obase[srow * D_MODEL + 16 * t + l15] = (__bf16)(o[t][r] * li);
  }
}

// ---------------------------------------------------------------------------
// Stage 3: output projection (bf16 x bf16 -> f32), 32x64 tile per wave,
// same ping-pong pipeline as stage 1.
// ---------------------------------------------------------------------------
__global__ __launch_bounds__(128) void out_gemm_kernel(
    const __bf16* __restrict__ X, const __bf16* __restrict__ W,
    const float* __restrict__ bias, float* __restrict__ out) {
  const int lane = threadIdx.x & 31;
  const int wave = threadIdx.x >> 5;
  const int gw   = blockIdx.x * 4 + wave;       // 4096 waves
  const int m0   = (gw >> 4) * 32;
  const int n0   = (gw & 15) * 64;
  const int half = lane >> 4;
  const int l15  = lane & 15;

  const __bf16* arow0 = X + (size_t)(m0 + l15) * D_MODEL;
  const __bf16* arow1 = arow0 + 16 * D_MODEL;
  const __bf16* wrow  = W + (size_t)(n0 + l15) * D_MODEL + 16 * half;

  v8f acc[2][4];
#pragma unroll
  for (int mi = 0; mi < 2; ++mi)
#pragma unroll
    for (int t = 0; t < 4; ++t) acc[mi][t] = v8f_zero();

  v16bf a0e, a1e, be[4];
  v16bf a0o, a1o, bo[4];
  gemm_load_chunk(arow0, arow1, wrow, half, 0, a0e, a1e, be);

  for (int k0 = 0; k0 < D_MODEL; k0 += 64) {
    __builtin_prefetch(arow0 + k0 + 64, 0, 3);
    __builtin_prefetch(arow1 + k0 + 64, 0, 3);
    gemm_load_chunk(arow0, arow1, wrow, half, k0 + 32, a0o, a1o, bo);
    gemm_compute_chunk(a0e, a1e, be, acc);
    if (k0 + 64 < D_MODEL)
      gemm_load_chunk(arow0, arow1, wrow, half, k0 + 64, a0e, a1e, be);
    gemm_compute_chunk(a0o, a1o, bo, acc);
  }

#pragma unroll
  for (int mi = 0; mi < 2; ++mi) {
#pragma unroll
    for (int t = 0; t < 4; ++t) {
      const int n = n0 + 16 * t + l15;
      const float bv = bias[n];
#pragma unroll
      for (int r = 0; r < 8; ++r)
        out[(size_t)(m0 + 16 * mi + r + 8 * half) * D_MODEL + n] =
            acc[mi][t][r] + bv;
    }
  }
}

// ---------------------------------------------------------------------------
extern "C" void kernel_launch(void* const* d_in, const int* in_sizes, int n_in,
                              void* d_out, int out_size, void* d_ws, size_t ws_size,
                              hipStream_t stream) {
  const float* dec = (const float*)d_in[0];
  const float* enc = (const float*)d_in[1];
  const float* Wq  = (const float*)d_in[2];
  const float* bq  = (const float*)d_in[3];
  const float* Wk  = (const float*)d_in[4];
  const float* bk  = (const float*)d_in[5];
  const float* Wv  = (const float*)d_in[6];
  const float* bv  = (const float*)d_in[7];
  const float* Wp  = (const float*)d_in[8];
  const float* bp  = (const float*)d_in[9];
  float* out = (float*)d_out;

  const size_t per = (size_t)BATCH * NHEAD * SEQ * HDIM;  // 8.39M elems
  const size_t wsz = (size_t)D_MODEL * D_MODEL;           // 1.05M elems
  __bf16* wsq  = (__bf16*)d_ws;        // [B,H,S,hd]
  __bf16* wsk  = wsq + per;            // [B,H,S,hd]
  __bf16* wsv  = wsk + per;            // [B,H,hd,S] (transposed)
  __bf16* wsa  = wsv + per;            // [B,S,D]    (attention output)
  __bf16* xdec = wsa + per;            // dec in bf16
  __bf16* xenc = xdec + per;           // enc in bf16
  __bf16* wq16 = xenc + per;           // weights in bf16
  __bf16* wk16 = wq16 + wsz;
  __bf16* wv16 = wk16 + wsz;
  __bf16* wp16 = wv16 + wsz;
  // total ws use: (6*per + 4*wsz) * 2 B ~= 109 MB

  dim3 blk(128);                        // 4 waves / block (wave32)
  dim3 gGemm(1024);                     // 4096 waves, 32x64 tiles
  dim3 gAttn(2048);                     // 8192 waves

  // Stage 0: one-time bf16 conversion (X and weights)
  cvt_bf16_kernel<<<dim3((int)(per / 2048)), dim3(256), 0, stream>>>(dec, xdec, (int)per);
  cvt_bf16_kernel<<<dim3((int)(per / 2048)), dim3(256), 0, stream>>>(enc, xenc, (int)per);
  cvt_bf16_kernel<<<dim3((int)(wsz / 2048)), dim3(256), 0, stream>>>(Wq, wq16, (int)wsz);
  cvt_bf16_kernel<<<dim3((int)(wsz / 2048)), dim3(256), 0, stream>>>(Wk, wk16, (int)wsz);
  cvt_bf16_kernel<<<dim3((int)(wsz / 2048)), dim3(256), 0, stream>>>(Wv, wv16, (int)wsz);
  cvt_bf16_kernel<<<dim3((int)(wsz / 2048)), dim3(256), 0, stream>>>(Wp, wp16, (int)wsz);

  // Stages 1-3
  proj_gemm_kernel<<<gGemm, blk, 0, stream>>>(xdec, wq16, bq, wsq, 0);
  proj_gemm_kernel<<<gGemm, blk, 0, stream>>>(xenc, wk16, bk, wsk, 0);
  proj_gemm_kernel<<<gGemm, blk, 0, stream>>>(xenc, wv16, bv, wsv, 1);
  attn_kernel<<<gAttn, blk, 0, stream>>>(wsq, wsk, wsv, wsa);
  out_gemm_kernel<<<gGemm, blk, 0, stream>>>(wsa, wp16, bp, out);
}
